// TransformerBlock_66992899883321
// MI455X (gfx1250) — matrix operand order repaired and checked by reading, not compiled, not run on hardware
//
#include <hip/hip_runtime.h>
#include <hip/hip_bf16.h>
#include <stdint.h>
#include <math.h>

// ---------------------------------------------------------------------------
// Problem constants (fixed by the reference)
// ---------------------------------------------------------------------------
constexpr int BB   = 2;
constexpr int SS   = 2048;
constexpr int DM   = 1024;   // model dim
constexpr int HH   = 16;     // heads
constexpr int DFFc = 4096;
constexpr int DKc  = 64;     // head dim

typedef __attribute__((ext_vector_type(16))) __bf16 v16bf;
typedef __attribute__((ext_vector_type(8)))  float  v8f;

#define DEV __device__ __forceinline__

// --------------------------- async copy to LDS ------------------------------
// gfx1250 GLOBAL_LOAD_ASYNC_TO_LDS_B128: per-lane 16B DMA global->LDS,
// tracked by ASYNCcnt. The builtin's parameter types (from the compiler
// diagnostic) are int-vector pointers: (v4i AS1*, v4i AS3*, imm offset, cpol).
#if defined(__AMDGCN__) && __has_builtin(__builtin_amdgcn_global_load_async_to_lds_b128)
#define HAVE_ASYNC_LDS 1
#else
#define HAVE_ASYNC_LDS 0
#endif

#if HAVE_ASYNC_LDS
typedef int v4i_vec __attribute__((vector_size(16)));
typedef __attribute__((address_space(1))) v4i_vec* g_v4i_p;
typedef __attribute__((address_space(3))) v4i_vec* l_v4i_p;

DEV void async_copy_b128(const void* g, void* l) {
  __builtin_amdgcn_global_load_async_to_lds_b128(
      (g_v4i_p)(uintptr_t)g, (l_v4i_p)(uintptr_t)l, 0, 0);
}
DEV void wait_async() {
#if __has_builtin(__builtin_amdgcn_s_wait_asynccnt)
  __builtin_amdgcn_s_wait_asynccnt(0);
#else
  asm volatile("s_wait_asynccnt 0" ::: "memory");
#endif
}
#endif

DEV v8f wmma_bf16(v16bf a, v16bf b, v8f c) {
  // D = A(16x32 bf16) * B(32x16 bf16) + C(16x16 f32)
  return __builtin_amdgcn_wmma_f32_16x16x32_bf16(
      /*neg_a=*/false, a, /*neg_b=*/false, b,
      /*c_mod=*/(short)0, c, /*reuse_a=*/false, /*reuse_b=*/false);
}

// A fragment (16x32, row-major source, leading dim `ld` in elements).
// Lane l holds row m=l&15; K elems [8*half,+8) in v0..3, [16+8*half,+8) in
// v4..7 (half=l>>4): two contiguous 16B runs.
DEV v16bf load_frag_a(const __bf16* base, int ld, int lane) {
  const int m = lane & 15, half = lane >> 4;
  const __bf16* p0 = base + (long)m * ld + 8 * half;
  const __bf16* p1 = p0 + 16;
  v16bf a;
#pragma unroll
  for (int i = 0; i < 8; ++i) { a[i] = p0[i]; a[8 + i] = p1[i]; }
  return a;
}

// B fragment (32x16, K-major source, leading dim `ld`).
// Lane l holds column N=l&15; K elems [8*half,+8) then [16+8*half,+8) (half=l>>4).
DEV v16bf load_frag_b(const __bf16* base, int ld, int lane) {
  const int n = lane & 15, half = lane >> 4;
  const __bf16* p = base + (long)(8 * half) * ld + n;
  v16bf b;
#pragma unroll
  for (int i = 0; i < 8; ++i) { b[i] = p[(long)i * ld]; b[8 + i] = p[(long)(16 + i) * ld]; }
  return b;
}

// ---------------------------------------------------------------------------
// Elementwise cast f32 -> bf16 (weights)
// ---------------------------------------------------------------------------
__global__ void cast_f32_bf16_kernel(const float* __restrict__ s,
                                     __bf16* __restrict__ d, long n) {
  long i = (long)blockIdx.x * blockDim.x + threadIdx.x;
  const long stride = (long)gridDim.x * blockDim.x;
  for (; i < n; i += stride) d[i] = (__bf16)s[i];
}

// ---------------------------------------------------------------------------
// LayerNorm over last dim (D=1024), block per row, bf16 output
// ---------------------------------------------------------------------------
__global__ __launch_bounds__(256)
void layernorm_bf16_kernel(const float* __restrict__ x,
                           const float* __restrict__ g,
                           const float* __restrict__ bta,
                           __bf16* __restrict__ out, int D) {
  __shared__ float red[256];
  const int t = threadIdx.x;
  const long row = blockIdx.x;
  const float* xr = x + row * D;
  float s = 0.f, s2 = 0.f;
  for (int i = t; i < D; i += 256) { float v = xr[i]; s += v; s2 += v * v; }
  red[t] = s; __syncthreads();
  for (int off = 128; off > 0; off >>= 1) {
    if (t < off) red[t] += red[t + off];
    __syncthreads();
  }
  const float mean = red[0] / (float)D;
  __syncthreads();
  red[t] = s2; __syncthreads();
  for (int off = 128; off > 0; off >>= 1) {
    if (t < off) red[t] += red[t + off];
    __syncthreads();
  }
  const float var = red[0] / (float)D - mean * mean;
  const float rstd = rsqrtf(var + 1e-5f);
  for (int i = t; i < D; i += 256) {
    float v = (xr[i] - mean) * rstd * g[i] + bta[i];
    out[row * D + i] = (__bf16)v;
  }
}

// ---------------------------------------------------------------------------
// Tiled WMMA GEMM: C = A(MxK bf16) @ B(KxN bf16) + bias.
// 128x128x32 block tile, 8 waves, wave tile 32x64 (2x4 WMMA tiles).
// Double-buffered LDS; tile fills use async global->LDS DMA when available,
// overlapped with the WMMAs of the current tile.
// EPI: 0 -> bf16 out; 1 -> f32 out + residual; 2 -> GELU, bf16 out
// ---------------------------------------------------------------------------
template <int EPI>
__global__ __launch_bounds__(256)
void gemm_bf16_kernel(const __bf16* __restrict__ A, const __bf16* __restrict__ Bm,
                      const float* __restrict__ bias, const float* __restrict__ res,
                      void* __restrict__ outp, int M, int N, int K) {
  __shared__ __bf16 sA[2][128 * 32];
  __shared__ __bf16 sB[2][32 * 128];
  const int t = threadIdx.x, lane = t & 31, w = t >> 5;
  const int wm = w >> 1, wn = w & 1;
  const long bm = (long)blockIdx.y * 128, bn = (long)blockIdx.x * 128;

  v8f acc[2][4];
#pragma unroll
  for (int mi = 0; mi < 2; ++mi)
#pragma unroll
    for (int ni = 0; ni < 4; ++ni)
#pragma unroll
      for (int r = 0; r < 8; ++r) acc[mi][ni][r] = 0.f;

  const int arow = t >> 1, ach = (t & 1) * 16;   // A tile: 128 rows x 32
  const int brow = t >> 3, bch = (t & 7) * 16;   // B tile: 32 rows x 128

  auto issue = [&](int k0, int buf) {
    const __bf16* ga = A + (bm + arow) * (long)K + k0 + ach;
    const __bf16* gb = Bm + (long)(k0 + brow) * N + bn + bch;
    __bf16* la = &sA[buf][arow * 32 + ach];
    __bf16* lb = &sB[buf][brow * 128 + bch];
#if HAVE_ASYNC_LDS
    async_copy_b128(ga, la);
    async_copy_b128(ga + 8, la + 8);
    async_copy_b128(gb, lb);
    async_copy_b128(gb + 8, lb + 8);
#else
    const uint4* g4a = reinterpret_cast<const uint4*>(ga);
    uint4* l4a = reinterpret_cast<uint4*>(la);
    l4a[0] = g4a[0]; l4a[1] = g4a[1];
    const uint4* g4b = reinterpret_cast<const uint4*>(gb);
    uint4* l4b = reinterpret_cast<uint4*>(lb);
    l4b[0] = g4b[0]; l4b[1] = g4b[1];
    if (k0 + 32 < K) {
      __builtin_prefetch(ga + 32, 0, 1);
      __builtin_prefetch(gb + (long)32 * N, 0, 1);
    }
#endif
  };

  issue(0, 0);
#if HAVE_ASYNC_LDS
  wait_async();
#endif
  __syncthreads();

  int buf = 0;
  for (int k0 = 0; k0 < K; k0 += 32) {
    const int nxt = buf ^ 1;
    const bool more = (k0 + 32 < K);
    if (more) issue(k0 + 32, nxt);   // fill next tile while computing this one

    v16bf aF[2], bF[4];
#pragma unroll
    for (int mi = 0; mi < 2; ++mi)
      aF[mi] = load_frag_a(&sA[buf][(wm * 32 + mi * 16) * 32], 32, lane);
#pragma unroll
    for (int ni = 0; ni < 4; ++ni)
      bF[ni] = load_frag_b(&sB[buf][wn * 64 + ni * 16], 128, lane);
#pragma unroll
    for (int mi = 0; mi < 2; ++mi)
#pragma unroll
      for (int ni = 0; ni < 4; ++ni)
        acc[mi][ni] = wmma_bf16(aF[mi], bF[ni], acc[mi][ni]);

    if (more) {
#if HAVE_ASYNC_LDS
      wait_async();                  // our DMA into `nxt` is done
#endif
      __syncthreads();               // everyone done reading `buf` / filling `nxt`
    }
    buf = nxt;
  }

  // Epilogue. C layout: lane l -> row = r + 8*(l>>4), col = l&15 per tile.
#pragma unroll
  for (int mi = 0; mi < 2; ++mi)
#pragma unroll
    for (int ni = 0; ni < 4; ++ni)
#pragma unroll
      for (int r = 0; r < 8; ++r) {
        const long row = bm + wm * 32 + mi * 16 + (lane >> 4) * 8 + r;
        const long col = bn + wn * 64 + ni * 16 + (lane & 15);
        const long idx = row * N + col;
        float v = acc[mi][ni][r] + bias[col];
        if (EPI == 0) {
          ((__bf16*)outp)[idx] = (__bf16)v;
        } else if (EPI == 1) {
          ((float*)outp)[idx] = v + res[idx];
        } else {
          float gl = 0.5f * v *
              (1.0f + tanhf(0.7978845608028654f * (v + 0.044715f * v * v * v)));
          ((__bf16*)outp)[idx] = (__bf16)gl;
        }
      }
}

// ---------------------------------------------------------------------------
// Attention pass 1: per (b, q, k) compute max/sum-exp of the masked+scaled
// scores across the 16 HEADS (reference softmax is over axis=1 = heads!).
// One wave per (b, 16q-tile, 16k-tile); 2 WMMAs per head (DK=64 = 2x K32).
// ---------------------------------------------------------------------------
__global__ __launch_bounds__(256)
void attn_stats_kernel(const __bf16* __restrict__ qb, const __bf16* __restrict__ kb,
                       const int* __restrict__ mask,
                       float* __restrict__ Mst, float* __restrict__ Zst) {
  __shared__ __bf16 kt[8][DKc * 16];   // per-wave K^T staging: [feat][key]
  const int t = threadIdx.x, lane = t & 31, w = t >> 5;
  const long wg = (long)blockIdx.x * 8 + w;
  const int nt = SS / 16;
  const long b = wg / ((long)nt * nt);
  const long rem = wg % ((long)nt * nt);
  const int qt = (int)(rem / nt), ktile = (int)(rem % nt);
  const int q0 = qt * 16, k0 = ktile * 16;
  if (b >= BB) return;

  const int mk = mask[b * SS + k0 + (lane & 15)];
  float m[8], z[8];
#pragma unroll
  for (int r = 0; r < 8; ++r) { m[r] = -1e30f; z[r] = 0.f; }

  __bf16* lkt = kt[w];
  for (int h = 0; h < HH; ++h) {
    // stage K^T tile: 64 feats x 16 keys (coalesced reads, transposed store)
    for (int e = lane; e < DKc * 16; e += 32) {
      const int key = e >> 6, f = e & 63;
      lkt[f * 16 + key] = kb[(b * SS + k0 + key) * (long)DM + h * DKc + f];
    }
    v8f acc;
#pragma unroll
    for (int r = 0; r < 8; ++r) acc[r] = 0.f;
#pragma unroll
    for (int c = 0; c < 2; ++c) {
      v16bf aQ = load_frag_a(qb + (b * SS + q0) * (long)DM + h * DKc + c * 32, DM, lane);
      v16bf bK = load_frag_b(lkt + (c * 32) * 16, 16, lane);
      acc = wmma_bf16(aQ, bK, acc);
    }
#pragma unroll
    for (int r = 0; r < 8; ++r) {
      const float sv = mk ? acc[r] * 0.125f : -1e9f;   // 1/sqrt(64), mask
      const float mn = fmaxf(m[r], sv);
      z[r] = z[r] * __expf(m[r] - mn) + __expf(sv - mn);
      m[r] = mn;
    }
  }
#pragma unroll
  for (int r = 0; r < 8; ++r) {
    const long q = q0 + (lane >> 4) * 8 + r;
    const long k = k0 + (lane & 15);
    Mst[(b * SS + q) * (long)SS + k] = m[r];
    Zst[(b * SS + q) * (long)SS + k] = z[r];
  }
}

// ---------------------------------------------------------------------------
// Attention pass 2: per (b, h, q-tile) recompute scores, normalize with the
// cross-head stats, and accumulate O = P @ V with WMMA. K loop in steps of 32.
// ---------------------------------------------------------------------------
__global__ __launch_bounds__(256)
void attn_pv_kernel(const __bf16* __restrict__ qb, const __bf16* __restrict__ kb,
                    const __bf16* __restrict__ vb, const int* __restrict__ mask,
                    const float* __restrict__ Mst, const float* __restrict__ Zst,
                    __bf16* __restrict__ ob) {
  __shared__ __bf16 kt[8][DKc * 32];  // per-wave K^T staging [feat][key]
  __shared__ __bf16 pt[8][16 * 32];   // per-wave P tile [q][k]
  const int t = threadIdx.x, lane = t & 31, w = t >> 5;
  const long wg = (long)blockIdx.x * 8 + w;
  const int nt = SS / 16;
  const long b = wg / ((long)HH * nt);
  const long rem = wg % ((long)HH * nt);
  const int h = (int)(rem / nt), qt = (int)(rem % nt);
  const int q0 = qt * 16;
  if (b >= BB) return;

  __bf16* lkt = kt[w];
  __bf16* lpt = pt[w];

  v16bf aQ[2];
#pragma unroll
  for (int c = 0; c < 2; ++c)
    aQ[c] = load_frag_a(qb + (b * SS + q0) * (long)DM + h * DKc + c * 32, DM, lane);

  v8f o[4];
#pragma unroll
  for (int ni = 0; ni < 4; ++ni)
#pragma unroll
    for (int r = 0; r < 8; ++r) o[ni][r] = 0.f;

  for (int k0 = 0; k0 < SS; k0 += 32) {
    // stage K^T: 64 feats x 32 keys
    for (int e = lane; e < DKc * 32; e += 32) {
      const int key = e >> 6, f = e & 63;
      lkt[f * 32 + key] = kb[(b * SS + k0 + key) * (long)DM + h * DKc + f];
    }
#pragma unroll
    for (int kt2 = 0; kt2 < 2; ++kt2) {
      v8f s;
#pragma unroll
      for (int r = 0; r < 8; ++r) s[r] = 0.f;
#pragma unroll
      for (int c = 0; c < 2; ++c) {
        v16bf bK = load_frag_b(lkt + (c * 32) * 32 + kt2 * 16, 32, lane);
        s = wmma_bf16(aQ[c], bK, s);
      }
#pragma unroll
      for (int r = 0; r < 8; ++r) {
        const int ql = (lane >> 4) * 8 + r;
        const int kl = kt2 * 16 + (lane & 15);
        const long kg = k0 + kl;
        const int mk = mask[b * SS + kg];
        const float sv = mk ? s[r] * 0.125f : -1e9f;
        const long qg = q0 + ql;
        const float Mv = Mst[(b * SS + qg) * (long)SS + kg];
        const float Zv = Zst[(b * SS + qg) * (long)SS + kg];
        lpt[ql * 32 + kl] = (__bf16)(__expf(sv - Mv) / Zv);
      }
    }
    // P (16x32) @ V (32x64): A from LDS, B straight from global (contiguous)
    v16bf aP = load_frag_a(lpt, 32, lane);
#pragma unroll
    for (int ni = 0; ni < 4; ++ni) {
      v16bf bV = load_frag_b(vb + (b * SS + k0) * (long)DM + h * DKc + ni * 16, DM, lane);
      o[ni] = wmma_bf16(aP, bV, o[ni]);
    }
  }

#pragma unroll
  for (int ni = 0; ni < 4; ++ni)
#pragma unroll
    for (int r = 0; r < 8; ++r) {
      const long row = q0 + (lane >> 4) * 8 + r;
      const long col = h * DKc + ni * 16 + (lane & 15);
      ob[(b * SS + row) * (long)DM + col] = (__bf16)o[ni][r];
    }
}

// ---------------------------------------------------------------------------
// Host-side launch
// ---------------------------------------------------------------------------
extern "C" void kernel_launch(void* const* d_in, const int* in_sizes, int n_in,
                              void* d_out, int out_size, void* d_ws, size_t ws_size,
                              hipStream_t stream) {
  (void)in_sizes; (void)n_in; (void)out_size; (void)ws_size;
  const long M = (long)BB * SS;

  const float* x    = (const float*)d_in[0];
  const int*   mask = (const int*)d_in[1];
  const float* ln1g = (const float*)d_in[2];
  const float* ln1b = (const float*)d_in[3];
  const float* Wq   = (const float*)d_in[4];
  const float* bq   = (const float*)d_in[5];
  const float* Wk   = (const float*)d_in[6];
  const float* bk   = (const float*)d_in[7];
  const float* Wv   = (const float*)d_in[8];
  const float* bv   = (const float*)d_in[9];
  const float* Wo   = (const float*)d_in[10];
  const float* bo   = (const float*)d_in[11];
  const float* ln2g = (const float*)d_in[12];
  const float* ln2b = (const float*)d_in[13];
  const float* W1   = (const float*)d_in[14];
  const float* b1   = (const float*)d_in[15];
  const float* W2   = (const float*)d_in[16];
  const float* b2   = (const float*)d_in[17];

  char* ws = (char*)d_ws;
  size_t off = 0;
  auto carve = [&](size_t bytes) -> void* {
    void* p = ws + off;
    off = (off + bytes + 255) & ~(size_t)255;
    return p;
  };

  __bf16* hb  = (__bf16*)carve(M * DM * 2);
  __bf16* qb  = (__bf16*)carve(M * DM * 2);
  __bf16* kb  = (__bf16*)carve(M * DM * 2);
  __bf16* vb  = (__bf16*)carve(M * DM * 2);
  __bf16* obf = (__bf16*)carve(M * DM * 2);
  __bf16* h2b = (__bf16*)carve(M * DM * 2);
  __bf16* a1b = (__bf16*)carve(M * DFFc * 2);
  __bf16* Wqb = (__bf16*)carve((long)DM * DM * 2);
  __bf16* Wkb = (__bf16*)carve((long)DM * DM * 2);
  __bf16* Wvb = (__bf16*)carve((long)DM * DM * 2);
  __bf16* Wob = (__bf16*)carve((long)DM * DM * 2);
  __bf16* W1b = (__bf16*)carve((long)DM * DFFc * 2);
  __bf16* W2b = (__bf16*)carve((long)DFFc * DM * 2);
  float*  Mst = (float*)carve((long)BB * SS * SS * 4);
  float*  Zst = (float*)carve((long)BB * SS * SS * 4);
  float*  x1  = (float*)carve(M * DM * 4);

  // --- cast weights to bf16 ---
  cast_f32_bf16_kernel<<<1024, 256, 0, stream>>>(Wq, Wqb, (long)DM * DM);
  cast_f32_bf16_kernel<<<1024, 256, 0, stream>>>(Wk, Wkb, (long)DM * DM);
  cast_f32_bf16_kernel<<<1024, 256, 0, stream>>>(Wv, Wvb, (long)DM * DM);
  cast_f32_bf16_kernel<<<1024, 256, 0, stream>>>(Wo, Wob, (long)DM * DM);
  cast_f32_bf16_kernel<<<1024, 256, 0, stream>>>(W1, W1b, (long)DM * DFFc);
  cast_f32_bf16_kernel<<<1024, 256, 0, stream>>>(W2, W2b, (long)DFFc * DM);

  // --- LN1 ---
  layernorm_bf16_kernel<<<(int)M, 256, 0, stream>>>(x, ln1g, ln1b, hb, DM);

  // --- QKV projections ---
  dim3 gD(DM / 128, (int)(M / 128));
  gemm_bf16_kernel<0><<<gD, 256, 0, stream>>>(hb, Wqb, bq, nullptr, qb, (int)M, DM, DM);
  gemm_bf16_kernel<0><<<gD, 256, 0, stream>>>(hb, Wkb, bk, nullptr, kb, (int)M, DM, DM);
  gemm_bf16_kernel<0><<<gD, 256, 0, stream>>>(hb, Wvb, bv, nullptr, vb, (int)M, DM, DM);

  // --- attention: cross-head softmax stats, then P@V ---
  attn_stats_kernel<<<(BB * (SS / 16) * (SS / 16)) / 8, 256, 0, stream>>>(
      qb, kb, mask, Mst, Zst);
  attn_pv_kernel<<<(BB * HH * (SS / 16)) / 8, 256, 0, stream>>>(
      qb, kb, vb, mask, Mst, Zst, obf);

  // --- output projection + residual ---
  gemm_bf16_kernel<1><<<gD, 256, 0, stream>>>(obf, Wob, bo, x, x1, (int)M, DM, DM);

  // --- LN2 ---
  layernorm_bf16_kernel<<<(int)M, 256, 0, stream>>>(x1, ln2g, ln2b, h2b, DM);

  // --- FFN ---
  dim3 gF(DFFc / 128, (int)(M / 128));
  gemm_bf16_kernel<2><<<gF, 256, 0, stream>>>(h2b, W1b, b1, nullptr, a1b, (int)M, DFFc, DM);
  gemm_bf16_kernel<1><<<gD, 256, 0, stream>>>(a1b, W2b, b2, x1, (float*)d_out, (int)M, DM, DFFc);
}
